// PatchReconstructor_49624052138408
// MI455X (gfx1250) — matrix operand order repaired and checked by reading, not compiled, not run on hardware
//
#include <hip/hip_runtime.h>

#define BASE_NUM 128
#define ROWS     588            // 196 patches * 3 channels
#define ROWS_PAD 592            // padded to multiple of 16
#define NCOL     256            // 16 x 16 base image
#define MT_FULL  36             // full 16-row output tiles (pairs: 18)
#define SCORES   (BASE_NUM * 42)     // 5376 floats per batch per score tensor
#define BASES    (BASE_NUM * 16)     // 2048 floats per batch per base tensor

typedef __attribute__((ext_vector_type(16))) __bf16 v16bf;
typedef __attribute__((ext_vector_type(8)))  __bf16 v8bf;
typedef __attribute__((ext_vector_type(8)))  float  v8f;

union BfFrag {
    v16bf v;
    uint4 u[2];
};
union Bf8 {
    v8bf v;
    uint4 u;
};

// LDS layout (dynamic LDS starts at offset 0 -> these are raw LDS addresses
// usable directly by the async-to-LDS instructions).
#define OFF_BT   0                                        // Bt  [256][128] bf16  (64 KB)
#define OFF_AS   (OFF_BT + NCOL * BASE_NUM * 2)           // As  [592][128] bf16  (148 KB)
#define RAW_HS   OFF_AS                                   // raw h_scores staging (consumed before As is written)
#define RAW_VS   (OFF_AS + SCORES * 4)                    // raw v_score staging
#define OFF_HT   (OFF_AS + ROWS_PAD * BASE_NUM * 2)       // hsT [42][128] f32    (21 KB)
#define OFF_VT   (OFF_HT + SCORES * 4)                    // vsT [42][128] f32    (21 KB)
#define OFF_HB   (OFF_VT + SCORES * 4)                    // hb  [128][16] f32    (8 KB)
#define OFF_VB   (OFF_HB + BASES * 4)                     // vb  [128][16] f32    (8 KB)
#define OFF_INV  (OFF_VB + BASES * 4)                     // inv [592] f32 row scale
#define LDS_BYTES (OFF_INV + ROWS_PAD * 4)                // 278848 B < 320 KB

// GLOBAL_LOAD_ASYNC_TO_LDS_B128 (GV mode): VDST = per-lane LDS byte offset,
// VADDR = per-lane 64-bit global address. Tracked by ASYNCcnt.
__device__ __forceinline__ void async_copy_b128(unsigned int lds_off, const void* gptr)
{
    asm volatile("global_load_async_to_lds_b128 %0, %1, off"
                 :: "v"(lds_off), "v"((unsigned long long)gptr)
                 : "memory");
}

extern "C" __global__ __launch_bounds__(256, 1)
void patch_reconstruct_wmma(const float* __restrict__ g_hs,
                            const float* __restrict__ g_vs,
                            const float* __restrict__ g_hb,
                            const float* __restrict__ g_vb,
                            float* __restrict__ g_out)
{
    extern __shared__ char smem[];
    __bf16* Bt  = (__bf16*)(smem + OFF_BT);
    __bf16* As  = (__bf16*)(smem + OFF_AS);
    float*  hsR = (float*)(smem + RAW_HS);
    float*  vsR = (float*)(smem + RAW_VS);
    float*  hsT = (float*)(smem + OFF_HT);
    float*  vsT = (float*)(smem + OFF_VT);
    float*  hb  = (float*)(smem + OFF_HB);
    float*  vb  = (float*)(smem + OFF_VB);
    float*  inv = (float*)(smem + OFF_INV);

    const int tid  = threadIdx.x;
    const int b    = blockIdx.x;
    const int lane = tid & 31;
    const int wave = tid >> 5;
    const int half = lane >> 4;              // 0: lanes 0..15 (K-low), 1: lanes 16..31 (K-high)
    const int l16  = lane & 15;

    // ---------- Phase 1a: async-stage per-batch inputs into LDS ----------
    {
        const float4* g0 = (const float4*)(g_hs + (size_t)b * SCORES);
        const float4* g1 = (const float4*)(g_vs + (size_t)b * SCORES);
        for (int i = tid; i < SCORES / 4; i += 256) {
            async_copy_b128(RAW_HS + i * 16, g0 + i);
            async_copy_b128(RAW_VS + i * 16, g1 + i);
        }
        const float4* g2 = (const float4*)(g_hb + (size_t)b * BASES);
        const float4* g3 = (const float4*)(g_vb + (size_t)b * BASES);
        for (int i = tid; i < BASES / 4; i += 256) {
            async_copy_b128(OFF_HB + i * 16, g2 + i);
            async_copy_b128(OFF_VB + i * 16, g3 + i);
        }
        asm volatile("s_wait_asynccnt 0" ::: "memory");
    }
    __syncthreads();

    // ---------- Phase 1b: transpose scores to K-contiguous + build B matrix ----------
    for (int i = tid; i < SCORES; i += 256) {
        const int k = i / 42, j = i - k * 42;
        hsT[j * BASE_NUM + k] = hsR[i];
        vsT[j * BASE_NUM + k] = vsR[i];
    }
    {
        const int j = tid;                   // 256 threads <-> 256 columns of B
        const float* vbm = vb + (j >> 4);
        const float* hbn = hb + (j & 15);
        for (int k = 0; k < BASE_NUM; k += 8) {
            Bf8 t;
            #pragma unroll
            for (int q = 0; q < 8; ++q)
                t.v[q] = (__bf16)(vbm[(k + q) * 16] * hbn[(k + q) * 16]);
            *(uint4*)(Bt + j * BASE_NUM + k) = t.u;
        }
    }
    __syncthreads();   // raw staging fully consumed; As region may now be overwritten

    // ---------- Phase 1c: softmax numerators (bf16) + per-row 1/sum ----------
    for (int r = tid; r < ROWS_PAD; r += 256) {
        if (r < ROWS) {
            const int c = r % 3, p = r / 3;
            const float4* hq = (const float4*)(hsT + (c * 14 + p / 14) * BASE_NUM);
            const float4* vq = (const float4*)(vsT + (c * 14 + p % 14) * BASE_NUM);
            float mx = -3.4e38f;
            #pragma unroll 4
            for (int k4 = 0; k4 < BASE_NUM / 4; ++k4) {
                const float4 h = hq[k4], v = vq[k4];
                mx = fmaxf(mx, fmaxf(fmaxf(h.x * v.x, h.y * v.y),
                                     fmaxf(h.z * v.z, h.w * v.w)));
            }
            float sum = 0.f;
            for (int k4 = 0; k4 < BASE_NUM / 4; k4 += 2) {
                const float4 h0 = hq[k4], v0 = vq[k4];
                const float4 h1 = hq[k4 + 1], v1 = vq[k4 + 1];
                float e[8];
                e[0] = __expf(h0.x * v0.x - mx); e[1] = __expf(h0.y * v0.y - mx);
                e[2] = __expf(h0.z * v0.z - mx); e[3] = __expf(h0.w * v0.w - mx);
                e[4] = __expf(h1.x * v1.x - mx); e[5] = __expf(h1.y * v1.y - mx);
                e[6] = __expf(h1.z * v1.z - mx); e[7] = __expf(h1.w * v1.w - mx);
                Bf8 t;
                #pragma unroll
                for (int q = 0; q < 8; ++q) { sum += e[q]; t.v[q] = (__bf16)e[q]; }
                *(uint4*)(As + r * BASE_NUM + k4 * 4) = t.u;
            }
            inv[r] = 1.0f / sum;              // normalization folded into phase-2 store
        } else {
            const uint4 z = {0u, 0u, 0u, 0u};
            for (int k = 0; k < BASE_NUM; k += 8)
                *(uint4*)(As + r * BASE_NUM + k) = z;
            inv[r] = 0.f;
        }
    }
    __syncthreads();

    // ---------- Phase 2: (592x128) @ (128x256) via v_wmma_f32_16x16x32_bf16 ----------
    // A 16-bit 16x32 fragment: lane%16 = M row; lanes 0-15 hold K 0-7 & 16-23, lanes 16-31 K 8-15 & 24-31.
    // B 16-bit 32x16 fragment: lane%16 = N col; lanes 0-15 hold K 0-7 & 16-23, lanes 16-31 K 8-15 & 24-31 (same as A).
    #define LOAD_A(aarr, mt)                                                     \
        {                                                                        \
            const int rbase_ = ((mt) * 16 + l16) * BASE_NUM;                     \
            _Pragma("unroll")                                                    \
            for (int kk = 0; kk < 4; ++kk) {                                     \
                const int k0_ = kk * 32 + half * 8;                              \
                aarr[kk].u[0] = *(const uint4*)(As + rbase_ + k0_);              \
                aarr[kk].u[1] = *(const uint4*)(As + rbase_ + k0_ + 16);         \
            }                                                                    \
        }
    #define LOAD_B(barr, nt)                                                     \
        {                                                                        \
            const __bf16* bcol_ = Bt + ((nt) * 16 + l16) * BASE_NUM + half * 8;  \
            _Pragma("unroll")                                                    \
            for (int kk = 0; kk < 4; ++kk) {                                     \
                barr[kk].u[0] = *(const uint4*)(bcol_ + kk * 32);                \
                barr[kk].u[1] = *(const uint4*)(bcol_ + kk * 32 + 16);           \
            }                                                                    \
        }
    #define LOAD_INV(rv, mt)                                                     \
        _Pragma("unroll")                                                        \
        for (int v = 0; v < 8; ++v) rv[v] = inv[(mt) * 16 + half * 8 + v];

    const size_t outBase = (size_t)b * ROWS * NCOL + (size_t)(half * 8) * NCOL + l16;

    // Full tiles, two M-tiles per iteration sharing each B fragment
    // (halves B LDS re-reads; two independent WMMA chains hide hazards).
    for (int mtp = wave; mtp < MT_FULL / 2; mtp += 8) {
        const int mt0 = mtp * 2, mt1 = mt0 + 1;
        BfFrag a0[4], a1[4];
        LOAD_A(a0, mt0);
        LOAD_A(a1, mt1);
        float rinv0[8], rinv1[8];
        LOAD_INV(rinv0, mt0);
        LOAD_INV(rinv1, mt1);
        float* outRow0 = g_out + outBase + (size_t)(mt0 * 16) * NCOL;
        float* outRow1 = g_out + outBase + (size_t)(mt1 * 16) * NCOL;

        BfFrag bfb[2][4];
        LOAD_B(bfb[0], 0);
        #pragma unroll 2
        for (int nt = 0; nt < 16; ++nt) {
            const int cur = nt & 1;
            LOAD_B(bfb[cur ^ 1], (nt + 1) & 15);   // prefetch next tile's B fragments
            v8f acc0 = {}, acc1 = {};
            #pragma unroll
            for (int kk = 0; kk < 4; ++kk) {
                acc0 = __builtin_amdgcn_wmma_f32_16x16x32_bf16(
                        false, a0[kk].v, false, bfb[cur][kk].v, (short)0, acc0, false, false);
                acc1 = __builtin_amdgcn_wmma_f32_16x16x32_bf16(
                        false, a1[kk].v, false, bfb[cur][kk].v, (short)0, acc1, false, false);
            }
            float* p0 = outRow0 + nt * 16;
            float* p1 = outRow1 + nt * 16;
            #pragma unroll
            for (int v = 0; v < 8; ++v) p0[(size_t)v * NCOL] = acc0[v] * rinv0[v];
            #pragma unroll
            for (int v = 0; v < 8; ++v) p1[(size_t)v * NCOL] = acc1[v] * rinv1[v];
        }
    }

    // Epilogue tile (rows 576..591; only 576..587 are real), owned by wave 7.
    if (wave == 7) {
        const int mt = MT_FULL;
        BfFrag a[4];
        LOAD_A(a, mt);
        float rinv[8];
        LOAD_INV(rinv, mt);
        float* outRow = g_out + outBase + (size_t)(mt * 16) * NCOL;

        for (int nt = 0; nt < 16; ++nt) {
            BfFrag bf[4];
            LOAD_B(bf, nt);
            v8f acc = {};
            #pragma unroll
            for (int kk = 0; kk < 4; ++kk)
                acc = __builtin_amdgcn_wmma_f32_16x16x32_bf16(
                        false, a[kk].v, false, bf[kk].v, (short)0, acc, false, false);
            float* po = outRow + nt * 16;
            #pragma unroll
            for (int v = 0; v < 4; ++v) po[(size_t)v * NCOL] = acc[v] * rinv[v];
            if (!half) {                       // rows 580..583 only
                #pragma unroll
                for (int v = 4; v < 8; ++v) po[(size_t)v * NCOL] = acc[v] * rinv[v];
            }
        }
    }
}

extern "C" void kernel_launch(void* const* d_in, const int* in_sizes, int n_in,
                              void* d_out, int out_size, void* d_ws, size_t ws_size,
                              hipStream_t stream)
{
    const float* h_scores = (const float*)d_in[0];
    const float* v_score  = (const float*)d_in[1];
    const float* h_base   = (const float*)d_in[2];
    const float* v_base   = (const float*)d_in[3];
    float* out = (float*)d_out;
    const int nb = in_sizes[0] / SCORES;     // batch count (128)
    patch_reconstruct_wmma<<<nb, 256, LDS_BYTES, stream>>>(h_scores, v_score, h_base, v_base, out);
}
